// MyHead_67242007986918
// MI455X (gfx1250) — matrix operand order repaired and checked by reading, not compiled, not run on hardware
//
#include <hip/hip_runtime.h>

// ---------------------------------------------------------------------------
// MI455X (gfx1250) implementation.
// Dominant cost: KPConv second einsum == GEMM [80000 x 3840] @ [3840 x 256]
// (~157 GFLOP) -> v_wmma_f32_16x16x32_f16 tiles (stage 3).
// The first einsum (wf = w^T @ nx, per-point 16x16x16) is ALSO mapped onto
// WMMA with the a-reduce zero-padded 16->32 (stage 2), removing the VALU
// bottleneck seen in round 1 (valu=705 vs wmma=30).
// Wt (f16, 2MB) and feats (f16, 41MB) are L2-resident (192MB L2).
// ---------------------------------------------------------------------------

typedef _Float16 v8h  __attribute__((ext_vector_type(8)));
typedef _Float16 v16h __attribute__((ext_vector_type(16)));
typedef float    v8f  __attribute__((ext_vector_type(8)));
typedef float    v4f  __attribute__((ext_vector_type(4)));

#define BATCH  4
#define NPTS   20000
#define TOTALP (BATCH * NPTS)   // 80000
#define CCH    256
#define NNB    16
#define KP     15
#define NCLS   17
#define GH     32
#define GW     128
#define KP_EXT 1.2f
#define BN_EPS 1e-5f

// Workspace layout (bytes)
#define FEATS_BYTES (size_t)(TOTALP * CCH * 2)          // 40,960,000  f16 feats
#define WT_BYTES    (size_t)(KP * CCH * CCH * 2)        //  1,966,080  f16 weights
#define OUT_BYTES   (size_t)(TOTALP * CCH * 4)          // 81,920,000  f32 pre-BN
#define OFF_WT      (FEATS_BYTES)
#define OFF_OUT     (OFF_WT + WT_BYTES)
#define OFF_SUMS    (OFF_OUT + OUT_BYTES)

// ---------------------------------------------------------------------------
// Kernel 0: kp_weights f32 [K][C][Cd] -> Wt f16 [K][d][c].  c is the WMMA
// reduce dim, contiguous, so a B fragment is one 32B contiguous lane load
// (this code's B order: lane = column N, 16 consecutive K halves per lane).
// ---------------------------------------------------------------------------
__global__ __launch_bounds__(256) void wt_transpose_kernel(
    const float* __restrict__ kpw, _Float16* __restrict__ Wt)
{
  int i = blockIdx.x * 256 + threadIdx.x;
  if (i >= KP * CCH * CCH) return;
  int k = i >> 16;
  int r = i & 0xFFFF;
  int d = r >> 8;
  int c = r & 0xFF;
  Wt[i] = (_Float16)kpw[((size_t)k << 16) + ((size_t)c << 8) + (size_t)d];
}

// ---------------------------------------------------------------------------
// Kernel 1: bilinear grid sample (border, align_corners=False) -> feats f16.
// One wave32 per point; lane l covers channels [8l, 8l+8). Coalesced.
// ---------------------------------------------------------------------------
__global__ __launch_bounds__(256) void sample_kernel(
    const float* __restrict__ x, const float* __restrict__ px,
    const float* __restrict__ py, _Float16* __restrict__ feats)
{
  int wave = threadIdx.x >> 5;
  int lane = threadIdx.x & 31;
  int p = blockIdx.x * 8 + wave;
  if (p >= TOTALP) return;
  int b = p / NPTS;

  float ix = fminf(fmaxf(((px[p] + 1.f) * (float)GW - 1.f) * 0.5f, 0.f), (float)(GW - 1));
  float iy = fminf(fmaxf(((py[p] + 1.f) * (float)GH - 1.f) * 0.5f, 0.f), (float)(GH - 1));
  float x0f = floorf(ix), y0f = floorf(iy);
  float wx = ix - x0f, wy = iy - y0f;
  int x0 = (int)x0f, y0 = (int)y0f;
  int x1 = min(x0 + 1, GW - 1), y1 = min(y0 + 1, GH - 1);

  const float* img = x + (size_t)b * (GH * GW) * CCH + lane * 8;
  const v4f* r00 = (const v4f*)(img + (size_t)(y0 * GW + x0) * CCH);
  const v4f* r01 = (const v4f*)(img + (size_t)(y0 * GW + x1) * CCH);
  const v4f* r10 = (const v4f*)(img + (size_t)(y1 * GW + x0) * CCH);
  const v4f* r11 = (const v4f*)(img + (size_t)(y1 * GW + x1) * CCH);
  float w00 = (1.f - wx) * (1.f - wy), w01 = wx * (1.f - wy);
  float w10 = (1.f - wx) * wy,         w11 = wx * wy;

  v8h o;
  #pragma unroll
  for (int j = 0; j < 2; ++j) {
    v4f a0 = r00[j], a1 = r01[j], a2 = r10[j], a3 = r11[j];
    #pragma unroll
    for (int e = 0; e < 4; ++e) {
      float v = a0[e] * w00 + a1[e] * w01 + a2[e] * w10 + a3[e] * w11;
      o[j * 4 + e] = (_Float16)v;
    }
  }
  *(v8h*)(feats + (size_t)p * CCH + lane * 8) = o;
}

// ---------------------------------------------------------------------------
// Kernel Z: zero the stats accumulators (512 floats).
// ---------------------------------------------------------------------------
__global__ void zero_kernel(float* __restrict__ sums)
{
  sums[threadIdx.x] = 0.f;
  sums[threadIdx.x + 256] = 0.f;
}

// ---------------------------------------------------------------------------
// Kernel 2: fused KPConv, all-WMMA core.
// One workgroup (256 thr = 8 wave32) per tile of 16 points.
//   geometry: w_h[n][k][a] f16 (k=15 row zeroed), once per tile
//   loop over 8 channel chunks of 32:
//     stage 1: transposing gather nx_s[n][c][a] (a-contig for B fragments;
//              a=16..31 pad zeroed once per tile)
//     stage 2: wf tile [k x c] per point via WMMA: A = w^T (16k x 32a,
//              hi-a = reg zeros), B = nx (32a x 16c), C = 0. 4 WMMA/wave.
//     stage 3: 15 k-steps x 2 d-tiles of WMMA per wave; A from LDS wf_s,
//              B from L2-resident Wt (32B contiguous lane loads),
//              global_prefetch of next-k B tile. 30 WMMA/wave.
// LDS: 8KB w_h + 32KB nx_s + 20.5KB wf_s + kpts = ~61.6 KB.
// ---------------------------------------------------------------------------
__global__ __launch_bounds__(256) void kpconv_kernel(
    const float* __restrict__ pxyz, const int* __restrict__ pknn,
    const float* __restrict__ kpts, const _Float16* __restrict__ feats,
    const _Float16* __restrict__ Wt, float* __restrict__ outp)
{
  __shared__ _Float16 w_h[16][16][16];   // [n][k][a], row k=15 zeroed
  __shared__ _Float16 nx_s[16][32][32];  // [n][c][a32], a>=16 zeroed (pad)
  __shared__ _Float16 wf_s[16][16][40];  // [k][n][c32], stride-40 rows
  __shared__ float    kpt_s[KP][3];

  int t  = threadIdx.x;
  int p0 = blockIdx.x * 16;              // tile never crosses a batch (20000%16==0)
  int b  = p0 / NPTS;
  int n  = t >> 4, a = t & 15;           // (point-in-tile, neighbor) worker
  int gp = p0 + n;
  int nb = b * NPTS + pknn[(size_t)gp * NNB + a];
  const _Float16* nbrow = feats + (size_t)nb * CCH;

  if (t < KP * 3) ((float*)kpt_s)[t] = kpts[t];

  // zero the a=16..31 pad of nx_s once (stage 1 never touches it)
  {
    v8h z = {};
    #pragma unroll
    for (int q = 0; q < 2; ++q) {
      int j = t * 2 + q;                 // 512 (n,c) pairs, 2 per thread
      v8h* dst = (v8h*)(&nx_s[j >> 5][j & 31][16]);
      dst[0] = z; dst[1] = z;
    }
  }

  float cx = pxyz[3 * gp],     cy = pxyz[3 * gp + 1], cz = pxyz[3 * gp + 2];
  float rx = pxyz[3 * nb] - cx, ry = pxyz[3 * nb + 1] - cy, rz = pxyz[3 * nb + 2] - cz;
  __syncthreads();                       // kpt_s visible
  #pragma unroll
  for (int k = 0; k < KP; ++k) {
    float dx = rx - kpt_s[k][0], dy = ry - kpt_s[k][1], dz = rz - kpt_s[k][2];
    float dist = sqrtf(dx * dx + dy * dy + dz * dz);
    w_h[n][k][a] = (_Float16)fmaxf(1.f - dist * (1.f / KP_EXT), 0.f);
  }
  w_h[n][15][a] = (_Float16)0.f;

  int wave   = t >> 5, lane = t & 31;
  int frow   = lane & 15;                // fragment row (M) / column (N) index
  int koff8  = (lane >> 4) * 8;          // A packing: halves {K0-7|K8-15},{K16-23|K24-31}
  int koff16 = (lane >> 4) * 16;         // B packing: 16 contiguous K halves per lane
  int mbase  = (lane >> 4) * 8;          // C/D packing: M = r + mbase
  int dt0    = wave * 2;                 // this wave's two 16-wide d-tiles
  int pn0    = wave * 2;                 // this wave's two stage-2 points
  v8f acc0 = {}, acc1 = {};
  v8h zero8 = {};

  __syncthreads();                       // w_h + nx pad ready

  for (int cb = 0; cb < 8; ++cb) {
    int cbase = cb * 32;

    // ---- stage 1: gather this thread's neighbor channels, transposed store
    {
      const v8h* src = (const v8h*)(nbrow + cbase);
      v8h h0 = src[0], h1 = src[1], h2 = src[2], h3 = src[3];
      #pragma unroll
      for (int e = 0; e < 8; ++e) {
        nx_s[n][e][a]      = h0[e];
        nx_s[n][8 + e][a]  = h1[e];
        nx_s[n][16 + e][a] = h2[e];
        nx_s[n][24 + e][a] = h3[e];
      }
    }
    __syncthreads();

    // ---- stage 2: wf[k][c] per point via WMMA (a-reduce padded 16->32)
    #pragma unroll
    for (int pq = 0; pq < 2; ++pq) {
      int pn = pn0 + pq;
      const v8h* wp = (const v8h*)(&w_h[pn][frow][koff8]);
      v16h A = __builtin_shufflevector(wp[0], zero8,
                 0,1,2,3,4,5,6,7,8,9,10,11,12,13,14,15);
      #pragma unroll
      for (int h = 0; h < 2; ++h) {
        int cloc = h * 16 + frow;
        const _Float16* bq = &nx_s[pn][cloc][koff8];
        v16h Bv = __builtin_shufflevector(*(const v8h*)bq, *(const v8h*)(bq + 16),
                   0,1,2,3,4,5,6,7,8,9,10,11,12,13,14,15);
        v8f d = {};
        d = __builtin_amdgcn_wmma_f32_16x16x32_f16(
                false, A, false, Bv, (short)0, d, false, false);
        #pragma unroll
        for (int r = 0; r < 8; ++r)
          wf_s[r + mbase][pn][cloc] = (_Float16)d[r];
      }
    }
    __syncthreads();

    // ---- stage 3: big GEMM accumulation over the 15 kernel points
    for (int k = 0; k < KP; ++k) {
      const _Float16* ap = &wf_s[k][frow][koff16];
      v8h alo = *(const v8h*)ap;
      v8h ahi = *(const v8h*)(ap + 8);
      v16h A = __builtin_shufflevector(alo, ahi,
                 0,1,2,3,4,5,6,7,8,9,10,11,12,13,14,15);

      const _Float16* bp0 = Wt +
          (((size_t)k * 256 + (size_t)(dt0 * 16 + frow)) * 256 + cbase + koff16);
      v16h B0 = *(const v16h*)bp0;
      acc0 = __builtin_amdgcn_wmma_f32_16x16x32_f16(
                 false, A, false, B0, (short)0, acc0, false, false);

      v16h B1 = *(const v16h*)(bp0 + 16 * 256);
      acc1 = __builtin_amdgcn_wmma_f32_16x16x32_f16(
                 false, A, false, B1, (short)0, acc1, false, false);

      // warm caches for next k's B tile (global_prefetch_b8)
      __builtin_prefetch((const void*)(bp0 + 65536), 0, 1);
    }
    __syncthreads();
  }

  // ---- epilogue: C/D layout -> rows: M = r + mbase, cols: lane%16
  int col = dt0 * 16 + frow;
  #pragma unroll
  for (int r = 0; r < 8; ++r) {
    size_t rowp = (size_t)(p0 + mbase + r) * CCH;
    outp[rowp + col] = acc0[r];
    outp[rowp + col + 16] = acc1[r];
  }
}

// ---------------------------------------------------------------------------
// Kernel 3: per-channel sum / sum-of-squares for training-mode BatchNorm.
// Thread = channel, block strides over rows (coalesced); f32 global atomics.
// ---------------------------------------------------------------------------
__global__ __launch_bounds__(256) void stats_kernel(
    const float* __restrict__ outp, float* __restrict__ sums)
{
  int c = threadIdx.x;
  float s = 0.f, q = 0.f;
  for (int row = blockIdx.x; row < TOTALP; row += gridDim.x) {
    float v = outp[(size_t)row * CCH + c];
    s += v;
    q += v * v;
  }
  atomicAdd(&sums[c], s);
  atomicAdd(&sums[256 + c], q);
}

// ---------------------------------------------------------------------------
// Kernel 4: BN (batch stats) + ReLU + 1x1 conv head, wave32 per point.
// Lane l holds channels [8l, 8l+8); 17 class dots reduced with __shfl_xor.
// ---------------------------------------------------------------------------
__global__ __launch_bounds__(256) void head_kernel(
    const float* __restrict__ outp, const float* __restrict__ sums,
    const float* __restrict__ gamma, const float* __restrict__ beta,
    const float* __restrict__ hw, const float* __restrict__ hb,
    float* __restrict__ logits)
{
  int wave = threadIdx.x >> 5;
  int lane = threadIdx.x & 31;
  int p = blockIdx.x * 8 + wave;
  if (p >= TOTALP) return;
  const float inv_n = 1.f / (float)TOTALP;
  int c0 = lane * 8;

  const v4f* rp = (const v4f*)(outp + (size_t)p * CCH + c0);
  v4f v0 = rp[0], v1 = rp[1];
  float h[8];
  #pragma unroll
  for (int e = 0; e < 8; ++e) {
    int c = c0 + e;
    float mean = sums[c] * inv_n;
    float var = sums[256 + c] * inv_n - mean * mean;
    float v = (e < 4) ? v0[e] : v1[e - 4];
    float hn = gamma[c] * (v - mean) * rsqrtf(var + BN_EPS) + beta[c];
    h[e] = fmaxf(hn, 0.f);
  }
  for (int cls = 0; cls < NCLS; ++cls) {
    const float* wr = hw + (size_t)cls * CCH + c0;
    float part = 0.f;
    #pragma unroll
    for (int e = 0; e < 8; ++e) part += h[e] * wr[e];
    for (int m = 16; m > 0; m >>= 1) part += __shfl_xor(part, m, 32);
    if (lane == 0) logits[(size_t)p * NCLS + cls] = part + hb[cls];
  }
}

// ---------------------------------------------------------------------------
extern "C" void kernel_launch(void* const* d_in, const int* in_sizes, int n_in,
                              void* d_out, int out_size, void* d_ws, size_t ws_size,
                              hipStream_t stream)
{
  const float* x     = (const float*)d_in[0];
  // d_in[1] = skip (unused: dead code in reference)
  const float* px    = (const float*)d_in[2];
  const float* py    = (const float*)d_in[3];
  const float* pxyz  = (const float*)d_in[4];
  const int*   pknn  = (const int*)d_in[5];
  // d_in[6] = num_points (unused)
  const float* kpts  = (const float*)d_in[7];
  const float* kpw   = (const float*)d_in[8];
  const float* gamma = (const float*)d_in[9];
  const float* beta  = (const float*)d_in[10];
  const float* hw    = (const float*)d_in[11];
  const float* hb    = (const float*)d_in[12];

  char* ws = (char*)d_ws;
  _Float16* feats = (_Float16*)(ws);
  _Float16* Wt    = (_Float16*)(ws + OFF_WT);
  float*    outp  = (float*)(ws + OFF_OUT);
  float*    sums  = (float*)(ws + OFF_SUMS);
  float*    logits = (float*)d_out;

  wt_transpose_kernel<<<(KP * CCH * CCH + 255) / 256, 256, 0, stream>>>(kpw, Wt);
  sample_kernel<<<TOTALP / 8, 256, 0, stream>>>(x, px, py, feats);
  zero_kernel<<<1, 256, 0, stream>>>(sums);
  kpconv_kernel<<<TOTALP / 16, 256, 0, stream>>>(pxyz, pknn, kpts, feats, Wt, outp);
  stats_kernel<<<400, 256, 0, stream>>>(outp, sums);
  head_kernel<<<TOTALP / 8, 256, 0, stream>>>(outp, sums, gamma, beta, hw, hb, logits);
}
